// efficient_40879498729144
// MI455X (gfx1250) — matrix operand order repaired and checked by reading, not compiled, not run on hardware
//
#include <hip/hip_runtime.h>
#include <hip/hip_bf16.h>

typedef __attribute__((ext_vector_type(16))) _Float16 v16h;
typedef __attribute__((ext_vector_type(8)))  _Float16 v8h;
typedef __attribute__((ext_vector_type(8)))  float    v8f;

#define BSZ   16
#define FE_C  256
#define FE_H  64
#define FE_W  64

__device__ __forceinline__ int iclamp(int v, int lo, int hi) {
    return v < lo ? lo : (v > hi ? hi : v);
}

// ---------------------------------------------------------------------------
// Correlate: depthwise 3x3 dynamic correlation, max over 3 objects.
// fe: [16][256][64][64] f32 (NCHW).  pr: [27][16][256] f32 (step slice).
// out: PADDED [16][66][66][256] f16 NHWC, zero halo written by this kernel.
// Proto weight (b,n,c,ky,kx) = pr[(n*9+ky*3+kx)*16*256 + b*256 + c]
// (wave-uniform per block -> scalar loads).
// ---------------------------------------------------------------------------
__global__ __launch_bounds__(256)
void correlate_kernel(const float* __restrict__ fe,
                      const float* __restrict__ pr,
                      _Float16* __restrict__ out)
{
    const int c = blockIdx.y;
    const int b = blockIdx.z;
    const int t = blockIdx.x * 256 + threadIdx.x;   // 0..66*66-1 (+pad)
    if (t >= 66 * 66) return;
    const int x = t % 66;
    const int y = t / 66;
    _Float16* op = out + (((size_t)b * 66 + y) * 66 + x) * FE_C + c;
    if (x == 0 || x == 65 || y == 0 || y == 65) { *op = (_Float16)0.f; return; }
    const int xs = x - 1, ys = y - 1;

    const float* f = fe + ((size_t)b * FE_C + c) * (FE_H * FE_W);
    float nb[9];
    #pragma unroll
    for (int ky = 0; ky < 3; ++ky) {
        const int yy = ys + ky - 1;
        #pragma unroll
        for (int kx = 0; kx < 3; ++kx) {
            const int xx = xs + kx - 1;
            const bool ok = (yy >= 0) && (yy < FE_H) && (xx >= 0) && (xx < FE_W);
            nb[ky * 3 + kx] = ok ? f[yy * FE_W + xx] : 0.f;
        }
    }
    float best = -3.4e38f;
    #pragma unroll
    for (int n = 0; n < 3; ++n) {
        float acc = 0.f;
        #pragma unroll
        for (int k = 0; k < 9; ++k)
            acc = fmaf(nb[k], pr[((n * 9 + k) * BSZ + b) * FE_C + c], acc);
        best = fmaxf(best, acc);
    }
    *op = (_Float16)best;
}

// ---------------------------------------------------------------------------
// Weight transform: [Cout][Cin][3][3] f32 -> [tap][Cout][Cin] f16.
// ---------------------------------------------------------------------------
__global__ __launch_bounds__(256)
void wtrans_kernel(const float* __restrict__ w, _Float16* __restrict__ wt,
                   int Cout, int Cin, int total)
{
    const int gid = blockIdx.x * 256 + threadIdx.x;
    if (gid >= total) return;
    const int ic  = gid % Cin;
    int r         = gid / Cin;
    const int oc  = r % Cout;
    const int tap = r / Cout;
    wt[gid] = (_Float16)w[((size_t)oc * Cin + ic) * 9 + tap];
}

// ---------------------------------------------------------------------------
// Implicit-GEMM 3x3 conv + bias + leaky ReLU via WMMA f16->f32.
// X: PADDED [16][H+2][W+2][CIN] f16 NHWC (zero halo -> no boundary checks).
// Wt: [9][COUT][CIN] f16.  Y: [16][H][W][COUT] f16 (unpadded).
// One wave: 32(oc) x 32(pixels) tile -> 2 A frags x 2 B frags feed 4 WMMAs
// (2-way reuse on both operands: 2 b128 loads per WMMA).
// A frag: lane<16 holds row M=l15, K {0..15}; lane>=16: K {16..31}.
// B frag: lane holds col N=l15, K-half = lane>>4 -> 16 contiguous channels.
// C/D:    VGPR v, lane<16 -> (M=v, N=lane); lane>=16 -> (M=v+8, N=lane-16).
// ---------------------------------------------------------------------------
template<int CIN, int COUT, int H, int W>
__global__ __launch_bounds__(128)
void conv3x3_wmma_kernel(const _Float16* __restrict__ X,
                         const _Float16* __restrict__ Wt,
                         const float* __restrict__ bias,
                         _Float16* __restrict__ Y)
{
    constexpr int NOC = COUT / 32;       // 32-channel oc tiles
    constexpr int TPR = W / 32;          // 32-pixel tiles per row
    constexpr int WP  = W + 2;           // padded row length
    const int wv   = (blockIdx.x * blockDim.x + threadIdx.x) >> 5;
    const int lane = threadIdx.x & 31;
    const int kh   = lane >> 4;          // K-half / M-half selector
    const int l15  = lane & 15;

    const int ocT = wv % NOC;
    int pixT      = wv / NOC;
    const int x0  = (pixT % TPR) * 32;
    pixT         /= TPR;
    const int y   = pixT % H;
    const int b   = pixT / H;
    const int oc0 = ocT * 32;

    v8f acc00 = {}, acc01 = {}, acc10 = {}, acc11 = {};
    const _Float16* wbase = Wt + (size_t)(oc0 + l15) * CIN + kh * 16;
    // output pixel (y, x0+l15) reads padded rows y..y+2, cols x0+l15 .. +2
    const _Float16* xbase = X + (((size_t)b * (H + 2) + y) * WP + (x0 + l15)) * CIN
                              + kh * 16;

    #pragma unroll
    for (int ky = 0; ky < 3; ++ky) {
        #pragma unroll
        for (int kx = 0; kx < 3; ++kx) {
            const _Float16* xrow = xbase + ((size_t)ky * WP + kx) * CIN;
            const _Float16* wrs  = wbase + (size_t)(ky * 3 + kx) * (COUT * CIN);
            #pragma unroll
            for (int ic = 0; ic < CIN; ic += 32) {
                // A fragment 0: output channels oc0 .. oc0+15
                v8h a0lo = *(const v8h*)(wrs + ic);
                v8h a0hi = *(const v8h*)(wrs + ic + 8);
                // A fragment 1: output channels oc0+16 .. oc0+31
                v8h a1lo = *(const v8h*)(wrs + (size_t)16 * CIN + ic);
                v8h a1hi = *(const v8h*)(wrs + (size_t)16 * CIN + ic + 8);
                v16h a0, a1;
                #pragma unroll
                for (int e = 0; e < 8; ++e) {
                    a0[e] = a0lo[e]; a0[8 + e] = a0hi[e];
                    a1[e] = a1lo[e]; a1[8 + e] = a1hi[e];
                }
                // B fragments: pixels x0+l15 and x0+16+l15
                const v16h b0 = *(const v16h*)(xrow + ic);
                const v16h b1 = *(const v16h*)(xrow + (size_t)16 * CIN + ic);
                acc00 = __builtin_amdgcn_wmma_f32_16x16x32_f16(
                            false, a0, false, b0, (short)0, acc00, false, false);
                acc01 = __builtin_amdgcn_wmma_f32_16x16x32_f16(
                            false, a0, false, b1, (short)0, acc01, false, false);
                acc10 = __builtin_amdgcn_wmma_f32_16x16x32_f16(
                            false, a1, false, b0, (short)0, acc10, false, false);
                acc11 = __builtin_amdgcn_wmma_f32_16x16x32_f16(
                            false, a1, false, b1, (short)0, acc11, false, false);
            }
        }
    }

    // bias + leaky ReLU (slope 0.01), convert, four b128 stores per lane
    const int mb = kh * 8;
    float bv0[8], bv1[8];
    #pragma unroll
    for (int v = 0; v < 8; ++v) {
        bv0[v] = bias[oc0 + mb + v];
        bv1[v] = bias[oc0 + 16 + mb + v];
    }
    v8h o00, o01, o10, o11;
    #pragma unroll
    for (int v = 0; v < 8; ++v) {
        float t00 = acc00[v] + bv0[v];
        float t01 = acc01[v] + bv0[v];
        float t10 = acc10[v] + bv1[v];
        float t11 = acc11[v] + bv1[v];
        t00 = (t00 > 0.f) ? t00 : 0.01f * t00;
        t01 = (t01 > 0.f) ? t01 : 0.01f * t01;
        t10 = (t10 > 0.f) ? t10 : 0.01f * t10;
        t11 = (t11 > 0.f) ? t11 : 0.01f * t11;
        o00[v] = (_Float16)t00;
        o01[v] = (_Float16)t01;
        o10[v] = (_Float16)t10;
        o11[v] = (_Float16)t11;
    }
    _Float16* yp = Y + (((size_t)b * H + y) * W + (x0 + l15)) * COUT + oc0 + mb;
    *(v8h*)yp = o00;                                   // px lo, oc lo
    *(v8h*)(yp + 16) = o10;                            // px lo, oc hi
    *(v8h*)(yp + (size_t)16 * COUT) = o01;             // px hi, oc lo
    *(v8h*)(yp + (size_t)16 * COUT + 16) = o11;        // px hi, oc hi
}

// ---------------------------------------------------------------------------
// Bilinear 2x upsample (half-pixel centers, edge clamp) on NHWC f16.
// in: [B][H][W][C] (unpadded conv output) -> out: [B][2H+2p][2W+2p][C];
// when pad=1, writes a zero halo so the next conv needs no boundary checks.
// ---------------------------------------------------------------------------
__global__ __launch_bounds__(256)
void up2_kernel(const _Float16* __restrict__ in, _Float16* __restrict__ out,
                int H, int W, int C, int pad, int total)
{
    const int gid = blockIdx.x * 256 + threadIdx.x;
    if (gid >= total) return;
    const int OW = 2 * W + 2 * pad;
    const int OH = 2 * H + 2 * pad;
    const int c  = gid % C;
    int r        = gid / C;
    const int xo = r % OW; r /= OW;
    const int yo = r % OH;
    const int b  = r / OH;

    if (pad && (xo == 0 || xo == OW - 1 || yo == 0 || yo == OH - 1)) {
        out[gid] = (_Float16)0.f;
        return;
    }
    const int xu = xo - pad;   // coordinate in the 2H x 2W upsampled image
    const int yu = yo - pad;

    const float sy = 0.5f * yu - 0.25f;
    const float fy = floorf(sy);
    const float wy = sy - fy;
    const int y0 = iclamp((int)fy,     0, H - 1);
    const int y1 = iclamp((int)fy + 1, 0, H - 1);
    const float sx = 0.5f * xu - 0.25f;
    const float fx = floorf(sx);
    const float wx = sx - fx;
    const int x0 = iclamp((int)fx,     0, W - 1);
    const int x1 = iclamp((int)fx + 1, 0, W - 1);

    const _Float16* base = in + (size_t)b * H * W * C;
    const float v00 = (float)base[((size_t)y0 * W + x0) * C + c];
    const float v01 = (float)base[((size_t)y0 * W + x1) * C + c];
    const float v10 = (float)base[((size_t)y1 * W + x0) * C + c];
    const float v11 = (float)base[((size_t)y1 * W + x1) * C + c];
    const float val = (1.f - wy) * ((1.f - wx) * v00 + wx * v01)
                    +        wy  * ((1.f - wx) * v10 + wx * v11);
    out[gid] = (_Float16)val;
}

// ---------------------------------------------------------------------------
// 1x1 conv 32->1 + ReLU.  X: [16][512][512][32] f16 NHWC.  out: f32.
// ---------------------------------------------------------------------------
__global__ __launch_bounds__(256)
void conv1x1_relu_kernel(const _Float16* __restrict__ X,
                         const float* __restrict__ w4,
                         const float* __restrict__ b4,
                         float* __restrict__ out, int total)
{
    const int gid = blockIdx.x * 256 + threadIdx.x;
    if (gid >= total) return;
    const _Float16* xp = X + (size_t)gid * 32;
    float acc = b4[0];
    #pragma unroll
    for (int c = 0; c < 32; ++c) acc = fmaf((float)xp[c], w4[c], acc);
    out[gid] = acc > 0.f ? acc : 0.f;
}

// ---------------------------------------------------------------------------
extern "C" void kernel_launch(void* const* d_in, const int* in_sizes, int n_in,
                              void* d_out, int out_size, void* d_ws, size_t ws_size,
                              hipStream_t stream)
{
    (void)in_sizes; (void)n_in; (void)out_size; (void)ws_size;

    const float* f_e  = (const float*)d_in[0];
    const float* prot = (const float*)d_in[1];   // [3][27][16][256]
    const float* w1   = (const float*)d_in[2];   // [3][128][256][3][3]
    const float* b1   = (const float*)d_in[3];   // [3][128]
    const float* w2   = (const float*)d_in[4];   // [3][64][128][3][3]
    const float* b2   = (const float*)d_in[5];   // [3][64]
    const float* w3   = (const float*)d_in[6];   // [3][32][64][3][3]
    const float* b3   = (const float*)d_in[7];   // [3][32]
    const float* w4   = (const float*)d_in[8];   // [3][32]
    const float* b4   = (const float*)d_in[9];   // [3][1]
    float* out = (float*)d_out;

    char* ws = (char*)d_ws;
    size_t off = 0;
    auto carve = [&](size_t bytes) -> char* {
        char* p = ws + off;
        off += (bytes + 255) & ~(size_t)255;
        return p;
    };
    _Float16* wt1  = (_Float16*)carve((size_t)9 * 128 * 256 * 2);
    _Float16* wt2  = (_Float16*)carve((size_t)9 * 64 * 128 * 2);
    _Float16* wt3  = (_Float16*)carve((size_t)9 * 32 * 64 * 2);
    // bufA holds padded conv inputs; largest use = conv4 input 16*512*512*32 f16
    _Float16* bufA = (_Float16*)carve((size_t)16 * 512 * 512 * 32 * 2); // 268 MB
    // bufB holds unpadded conv outputs; largest = conv3 out 16*256*256*32 f16
    _Float16* bufB = (_Float16*)carve((size_t)16 * 256 * 256 * 32 * 2); //  67 MB

    const size_t DM = (size_t)16 * 512 * 512;   // one density map (f32 elems)

    for (int i = 0; i < 3; ++i) {
        // return order: (outputs[2], (outputs[0], outputs[1]))
        float* dmap = out + ((i == 2) ? 0 : (size_t)(i + 1) * DM);

        // weight transforms for this step
        {   const int tot = 128 * 256 * 9;
            wtrans_kernel<<<(tot + 255) / 256, 256, 0, stream>>>(
                w1 + (size_t)i * tot, wt1, 128, 256, tot); }
        {   const int tot = 64 * 128 * 9;
            wtrans_kernel<<<(tot + 255) / 256, 256, 0, stream>>>(
                w2 + (size_t)i * tot, wt2, 64, 128, tot); }
        {   const int tot = 32 * 64 * 9;
            wtrans_kernel<<<(tot + 255) / 256, 256, 0, stream>>>(
                w3 + (size_t)i * tot, wt3, 32, 64, tot); }

        // correlate -> bufA padded [16][66][66][256] f16 (halo zeroed here)
        correlate_kernel<<<dim3((66 * 66 + 255) / 256, 256, 16), 256, 0, stream>>>(
            f_e, prot + (size_t)i * 27 * 16 * 256, bufA);

        // conv1: 256->128 @64x64 (WMMA, 32x32 tile/wave) -> bufB
        {   const int waves = (128 / 32) * (16 * 64 * 64 / 32);
            conv3x3_wmma_kernel<256, 128, 64, 64><<<waves / 4, 128, 0, stream>>>(
                bufA, wt1, b1 + (size_t)i * 128, bufB); }
        // up to padded [16][130][130][128] -> bufA
        {   const int tot = 16 * 130 * 130 * 128;
            up2_kernel<<<(tot + 255) / 256, 256, 0, stream>>>(bufB, bufA, 64, 64, 128, 1, tot); }
        // conv2: 128->64 @128x128 (WMMA) -> bufB
        {   const int waves = (64 / 32) * (16 * 128 * 128 / 32);
            conv3x3_wmma_kernel<128, 64, 128, 128><<<waves / 4, 128, 0, stream>>>(
                bufA, wt2, b2 + (size_t)i * 64, bufB); }
        // up to padded [16][258][258][64] -> bufA
        {   const int tot = 16 * 258 * 258 * 64;
            up2_kernel<<<(tot + 255) / 256, 256, 0, stream>>>(bufB, bufA, 128, 128, 64, 1, tot); }
        // conv3: 64->32 @256x256 (WMMA) -> bufB
        {   const int waves = (32 / 32) * (16 * 256 * 256 / 32);
            conv3x3_wmma_kernel<64, 32, 256, 256><<<waves / 4, 128, 0, stream>>>(
                bufA, wt3, b3 + (size_t)i * 32, bufB); }
        // up to unpadded [16][512][512][32] -> bufA
        {   const int tot = 16 * 512 * 512 * 32;
            up2_kernel<<<(tot + 255) / 256, 256, 0, stream>>>(bufB, bufA, 256, 256, 32, 0, tot); }
        // conv4: 1x1 32->1 + ReLU -> dmap (f32)
        conv1x1_relu_kernel<<<((int)DM + 255) / 256, 256, 0, stream>>>(
            bufA, w4 + (size_t)i * 32, b4 + i, dmap, (int)DM);
    }
}
